// BiLSTM_CRF_36309653520932
// MI455X (gfx1250) — matrix operand order repaired and checked by reading, not compiled, not run on hardware
//
#include <hip/hip_runtime.h>
#include <hip/hip_bf16.h>

// ---------------------------------------------------------------------------
// BiLSTM-CRF for MI455X (gfx1250, wave32, WMMA).
//
// All GEMMs (x@Wih^T, h@Whh^T, h@Wtag^T) use v_wmma_f32_16x16x32_bf16:
// bf16 inputs, f32 accumulation. Per-wave macro-tiles of 32x64 (2x4 WMMA
// tiles) give 8 WMMAs per 12 b128 loads (A reused 4x, B reused 2x).
// The recurrence runs one workgroup per direction, all 512 timesteps in a
// loop, with h(bf16)/c(f32)/gates(f32) resident in LDS (~176KB < 320KB WGP).
// Whh is deliberately re-streamed from L2 every timestep; an opaque ZERO
// OFFSET (not an opaque pointer) blocks cross-timestep hoisting while
// preserving the global address space (global_load, LOADcnt-only waits).
// ---------------------------------------------------------------------------

#define DEV __device__ __forceinline__

typedef __attribute__((ext_vector_type(16))) __bf16 bf16x16;
typedef __attribute__((ext_vector_type(8)))  __bf16 bf16x8;
typedef __attribute__((ext_vector_type(8)))  float  v8f;

static constexpr int B_ = 32, T_ = 512, EMB_ = 256, H_ = 256, HID_ = 512;
static constexpr int G4_ = 1024, L_ = 32, M_ = B_ * T_;

DEV __bf16 f2bf(float f) {
  union { float f; unsigned u; } x; x.f = f;
  unsigned r = x.u + 0x7FFFu + ((x.u >> 16) & 1u);   // round-to-nearest-even
  unsigned short h = (unsigned short)(r >> 16);
  __bf16 b; __builtin_memcpy(&b, &h, 2); return b;
}

DEV float sigm(float x) { return 1.f / (1.f + __expf(-x)); }

// A fragment (16x32 bf16, MxK). Per ISA layout: lane(hi,m), element e:
//   k = 16*(e>>3) + 8*hi + (e&7)  -> two contiguous 16B chunks per lane.
DEV bf16x16 load_a_frag(const __bf16* X, int ldx, int m0, int k0, int lane) {
  const int hi = lane >> 4;
  const __bf16* p = X + (long)(m0 + (lane & 15)) * ldx + k0;
  bf16x8 lo  = *(const bf16x8*)(p + 8 * hi);
  bf16x8 hi8 = *(const bf16x8*)(p + 16 + 8 * hi);
  bf16x16 a;
#pragma unroll
  for (int i = 0; i < 8; ++i) { a[i] = lo[i]; a[i + 8] = hi8[i]; }
  return a;
}

// B fragment (32x16 bf16, KxN), W stored row-major (N,K) so B[k][n]=W[n][k].
// Per ISA layout: lane(hi,n), element e: k = 16*(e>>3) + 8*hi + (e&7) -> two 16B chunks.
DEV bf16x16 load_b_frag(const __bf16* W, int ldw, int n0, int k0, int lane) {
  const int hi = lane >> 4;
  const __bf16* p = W + (long)(n0 + (lane & 15)) * ldw + k0 + 8 * hi;
  bf16x8 lo  = *(const bf16x8*)(p);
  bf16x8 hi8 = *(const bf16x8*)(p + 16);
  bf16x16 b;
#pragma unroll
  for (int i = 0; i < 8; ++i) { b[i] = lo[i]; b[i + 8] = hi8[i]; }
  return b;
}

DEV v8f wmma_bf16(bf16x16 a, bf16x16 b, v8f c) {
  return __builtin_amdgcn_wmma_f32_16x16x32_bf16(
      /*neg_a=*/false, a, /*neg_b=*/false, b,
      /*c_mod=*/(short)0, c, /*reuse_a=*/false, /*reuse_b=*/false);
}

// ---------------------------------------------------------------------------
// Small utility kernels
// ---------------------------------------------------------------------------
__global__ void k_f32_to_bf16(const float* __restrict__ s, __bf16* __restrict__ d, long n) {
  for (long i = blockIdx.x * (long)blockDim.x + threadIdx.x; i < n;
       i += (long)gridDim.x * blockDim.x)
    d[i] = f2bf(s[i]);
}

__global__ void k_bias_sum(const float* __restrict__ a, const float* __restrict__ b,
                           float* __restrict__ o, int n) {
  int i = blockIdx.x * blockDim.x + threadIdx.x;
  if (i < n) o[i] = a[i] + b[i];
}

__global__ void k_embed(const int* __restrict__ ids, const float* __restrict__ emb,
                        __bf16* __restrict__ x0) {
  int row = blockIdx.x;           // token index in [0, B*T)
  int e   = threadIdx.x;          // EMB = 256
  x0[(long)row * EMB_ + e] = f2bf(emb[(long)ids[row] * EMB_ + e]);
}

// ---------------------------------------------------------------------------
// Generic WMMA GEMM: Out[M,N] = X[M,K](bf16) @ W[N,K]^T(bf16) + bias[N]
// 8 waves per block; each wave owns a (16*MT) x (16*NT) macro-tile:
// MT*NT accumulators, A fragments reused NT times, B fragments MT times.
// Exit guard is wave-uniform so EXEC is all-ones at every WMMA.
// ---------------------------------------------------------------------------
template <int MT, int NT>
__global__ void __launch_bounds__(256)
k_wmma_gemm_bias(const __bf16* __restrict__ X, const __bf16* __restrict__ W,
                 const float* __restrict__ bias, float* __restrict__ Out,
                 int M, int N, int K) {
  const int lane = threadIdx.x & 31;
  const int wave = threadIdx.x >> 5;
  const int tn = N / (16 * NT);
  const long tile = (long)blockIdx.x * 8 + wave;
  const long tiles = (long)(M / (16 * MT)) * tn;
  if (tile >= tiles) return;                      // wave-uniform
  const int m0 = (int)(tile / tn) * 16 * MT;
  const int n0 = (int)(tile % tn) * 16 * NT;

  v8f acc[MT][NT];
#pragma unroll
  for (int mt = 0; mt < MT; ++mt)
#pragma unroll
    for (int nt = 0; nt < NT; ++nt) acc[mt][nt] = v8f{};

  for (int k0 = 0; k0 < K; k0 += 32) {
    __builtin_prefetch(X + (long)m0 * K + k0 + 128, 0, 3);  // RT-scope prefetch
    bf16x16 a[MT];
#pragma unroll
    for (int mt = 0; mt < MT; ++mt)
      a[mt] = load_a_frag(X, K, m0 + 16 * mt, k0, lane);
#pragma unroll
    for (int nt = 0; nt < NT; ++nt) {
      bf16x16 b = load_b_frag(W, K, n0 + 16 * nt, k0, lane);
#pragma unroll
      for (int mt = 0; mt < MT; ++mt)
        acc[mt][nt] = wmma_bf16(a[mt], b, acc[mt][nt]);
    }
  }
  const int hi = lane >> 4, nl = lane & 15;
#pragma unroll
  for (int nt = 0; nt < NT; ++nt) {
    const int n = n0 + 16 * nt + nl;
    const float bn = bias ? bias[n] : 0.f;
#pragma unroll
    for (int mt = 0; mt < MT; ++mt)
#pragma unroll
      for (int r = 0; r < 8; ++r)
        Out[(long)(m0 + 16 * mt + r + 8 * hi) * N + n] = acc[mt][nt][r] + bn; // m=r+8*hi
  }
}

// ---------------------------------------------------------------------------
// LSTM recurrence: one workgroup per direction (blockIdx.x: 0=fwd, 1=bwd).
// Per step: G(32x1024) = h_bf16(32x256) @ Whh^T via WMMA -> LDS, then gates.
// Wave w owns n-tiles [w*8, w*8+8), processed as 2 groups of 4 (acc[8] live)
// to keep VGPR pressure ~100 and avoid scratch spills.
// LDS: G 128KB + c 32KB + h_bf16 16KB = 176KB (dynamic).
// ---------------------------------------------------------------------------
__global__ void __launch_bounds__(256)
k_lstm_rec(const float* __restrict__ xg_f, const float* __restrict__ xg_b,
           const __bf16* __restrict__ whh_f, const __bf16* __restrict__ whh_b,
           __bf16* __restrict__ act, int T) {
  extern __shared__ char smem[];
  float*  G   = (float*)smem;                                   // [32][1024]
  float*  Cst = (float*)(smem + 32 * 1024 * 4);                 // [32][256]
  __bf16* Hbf = (__bf16*)(smem + 32 * 1024 * 4 + 32 * 256 * 4); // [32][256]

  const int dir  = blockIdx.x;
  const int lane = threadIdx.x & 31;
  const int wave = threadIdx.x >> 5;
  const float*  xg  = dir ? xg_b : xg_f;
  const __bf16* Whh = dir ? whh_b : whh_f;

  for (int u = threadIdx.x; u < 32 * 256; u += 256) { Cst[u] = 0.f; Hbf[u] = f2bf(0.f); }
  __syncthreads();

  for (int s = 0; s < T; ++s) {
    const int t = dir ? (T - 1 - s) : s;

    // Opaque zero offset: blocks cross-timestep hoist/spill of the
    // t-invariant Whh fragments, but keeps the pointer's global address
    // space so loads stay global_load (LOADcnt only), not flat_load.
    long opq = 0;
    asm volatile("" : "+s"(opq));
    const __bf16* WhhT = Whh + opq;

    // --- GEMM phase: 2 groups of 4 n-tiles, 2 m-tiles each ---
#pragma unroll 1
    for (int g = 0; g < 2; ++g) {
      v8f acc[8];
#pragma unroll
      for (int i = 0; i < 8; ++i) acc[i] = v8f{};
#pragma unroll
      for (int k0 = 0; k0 < 256; k0 += 32) {
        bf16x16 a0 = load_a_frag(Hbf, 256, 0,  k0, lane);   // LDS (ds_read)
        bf16x16 a1 = load_a_frag(Hbf, 256, 16, k0, lane);
#pragma unroll
        for (int nt = 0; nt < 4; ++nt) {
          const int n0 = ((wave << 3) + (g << 2) + nt) << 4;
          bf16x16 bf = load_b_frag(WhhT, 256, n0, k0, lane); // L2-resident
          acc[2 * nt]     = wmma_bf16(a0, bf, acc[2 * nt]);
          acc[2 * nt + 1] = wmma_bf16(a1, bf, acc[2 * nt + 1]);
        }
      }
      const int hi = lane >> 4, nl = lane & 15;
#pragma unroll
      for (int nt = 0; nt < 4; ++nt) {
        const int n = (((wave << 3) + (g << 2) + nt) << 4) + nl;
#pragma unroll
        for (int r = 0; r < 8; ++r) {
          G[(r + 8 * hi) * 1024 + n]      = acc[2 * nt][r];
          G[(16 + r + 8 * hi) * 1024 + n] = acc[2 * nt + 1][r];
        }
      }
    }
    __syncthreads();

    // --- Gate phase: thread tid handles hidden unit j=tid for batch b ---
    for (int u = threadIdx.x; u < 32 * 256; u += 256) {
      const int b = u >> 8, j = u & 255;
      const long row = ((long)b * T + t) * 1024;
      const float gi = G[b * 1024 + j]        + xg[row + j];
      const float gf = G[b * 1024 + 256 + j]  + xg[row + 256 + j];
      const float gg = G[b * 1024 + 512 + j]  + xg[row + 512 + j];
      const float go = G[b * 1024 + 768 + j]  + xg[row + 768 + j];
      const float c  = sigm(gf) * Cst[u] + sigm(gi) * tanhf(gg);
      const float h  = sigm(go) * tanhf(c);
      Cst[u] = c;
      const __bf16 hb = f2bf(h);
      Hbf[u] = hb;
      act[((long)b * T + t) * HID_ + (dir << 8) + j] = hb;
    }
    __syncthreads();
  }
}

// ---------------------------------------------------------------------------
// CRF forward algorithm: 1 block, 1024 threads; wave per batch (lane = label).
// ---------------------------------------------------------------------------
__global__ void k_crf_forward(const float* __restrict__ emis, const float* __restrict__ mask,
                              const float* __restrict__ trans, const float* __restrict__ startt,
                              const float* __restrict__ endt, float* __restrict__ logZ, int T) {
  __shared__ float alpha[32][32];
  __shared__ float tr[32][32];
  const int tid = threadIdx.x;
  const int b = tid >> 5, j = tid & 31;
  tr[b][j] = trans[tid];                      // tr[i][j], i = tid>>5
  alpha[b][j] = startt[j] + emis[(long)b * T * L_ + j];
  __syncthreads();
  for (int t = 1; t < T; ++t) {
    const float m = mask[b * T + t];
    const float e = emis[((long)b * T + t) * L_ + j];
    float mx = -3.4e38f;
#pragma unroll
    for (int i = 0; i < 32; ++i) mx = fmaxf(mx, alpha[b][i] + tr[i][j]);
    float s = 0.f;
#pragma unroll
    for (int i = 0; i < 32; ++i) s += __expf(alpha[b][i] + tr[i][j] - mx);
    const float nv = mx + __logf(s) + e;
    const float old = alpha[b][j];
    __syncthreads();
    alpha[b][j] = (m > 0.f) ? nv : old;
    __syncthreads();
  }
  float v = alpha[b][j] + endt[j];            // wave32 logsumexp over lanes
  float mx = v;
  for (int off = 16; off; off >>= 1) mx = fmaxf(mx, __shfl_xor(mx, off, 32));
  float s = __expf(v - mx);
  for (int off = 16; off; off >>= 1) s += __shfl_xor(s, off, 32);
  if (j == 0) logZ[b] = mx + __logf(s);
}

__global__ void k_gold(const float* __restrict__ emis, const int* __restrict__ labels,
                       const float* __restrict__ mask, const float* __restrict__ trans,
                       const float* __restrict__ startt, const float* __restrict__ endt,
                       float* __restrict__ gold, int T) {
  const int b = threadIdx.x;
  if (b >= 32) return;
  const int* tg = labels + (long)b * T;
  float sc = startt[tg[0]] + emis[(long)b * T * L_ + tg[0]];
  float msum = mask[b * T];
  for (int t = 1; t < T; ++t) {
    const float m = mask[b * T + t];
    msum += m;
    sc += (trans[tg[t] * L_ + tg[t - 1]] + emis[((long)b * T + t) * L_ + tg[t]]) * m;
  }
  sc += endt[tg[(int)msum - 1]];
  gold[b] = sc;
}

__global__ void k_viterbi(const float* __restrict__ emis, const float* __restrict__ mask,
                          const float* __restrict__ trans, const float* __restrict__ startt,
                          const float* __restrict__ endt, unsigned char* __restrict__ bp,
                          int* __restrict__ last_tag, int T) {
  __shared__ float V[32][32];
  __shared__ float tr[32][32];
  const int tid = threadIdx.x;
  const int b = tid >> 5, j = tid & 31;
  tr[b][j] = trans[tid];
  V[b][j] = startt[j] + emis[(long)b * T * L_ + j];
  __syncthreads();
  for (int t = 1; t < T; ++t) {
    const float m = mask[b * T + t];
    const float e = emis[((long)b * T + t) * L_ + j];
    float best = -3.4e38f; int bi = 0;
#pragma unroll
    for (int i = 0; i < 32; ++i) {
      const float s = V[b][i] + tr[i][j];
      if (s > best) { best = s; bi = i; }     // first max == jnp.argmax
    }
    bp[((long)b * T + t) * L_ + j] = (unsigned char)bi;
    const float nv = best + e;
    const float old = V[b][j];
    __syncthreads();
    V[b][j] = (m > 0.f) ? nv : old;
    __syncthreads();
  }
  float v = V[b][j] + endt[j];
  int bi = j;
  for (int off = 16; off; off >>= 1) {
    const float ov = __shfl_xor(v, off, 32);
    const int   oi = __shfl_xor(bi, off, 32);
    if (ov > v || (ov == v && oi < bi)) { v = ov; bi = oi; }
  }
  if (j == 0) last_tag[b] = bi;
}

__global__ void k_backtrack_loss(const unsigned char* __restrict__ bp,
                                 const int* __restrict__ last_tag,
                                 const float* __restrict__ logZ, const float* __restrict__ gold,
                                 float* __restrict__ out, int T) {
  const int b = threadIdx.x;
  if (b < 32) {
    int tag = last_tag[b];
    out[1 + (long)b * T + (T - 1)] = (float)tag;
    for (int t = T - 1; t >= 1; --t) {
      tag = bp[((long)b * T + t) * L_ + tag];
      out[1 + (long)b * T + (t - 1)] = (float)tag;
    }
  }
  if (threadIdx.x == 0) {
    float s = 0.f;
    for (int i = 0; i < 32; ++i) s += logZ[i] - gold[i];
    out[0] = s / 32.f;
  }
}

// ---------------------------------------------------------------------------
// Host launch
// ---------------------------------------------------------------------------
extern "C" void kernel_launch(void* const* d_in, const int* in_sizes, int n_in,
                              void* d_out, int out_size, void* d_ws, size_t ws_size,
                              hipStream_t stream) {
  (void)in_sizes; (void)n_in; (void)out_size; (void)ws_size;
  // Input order (setup_inputs dict, lstm_params flattened l0d0..l1d1):
  const int*   ids    = (const int*)  d_in[0];
  const float* mask   = (const float*)d_in[1];
  const int*   labels = (const int*)  d_in[2];
  const float* embed  = (const float*)d_in[3];
  const float* wih[4] = { (const float*)d_in[4],  (const float*)d_in[8],
                          (const float*)d_in[12], (const float*)d_in[16] }; // l0d0,l0d1,l1d0,l1d1
  const float* whh[4] = { (const float*)d_in[5],  (const float*)d_in[9],
                          (const float*)d_in[13], (const float*)d_in[17] };
  const float* bih[4] = { (const float*)d_in[6],  (const float*)d_in[10],
                          (const float*)d_in[14], (const float*)d_in[18] };
  const float* bhh[4] = { (const float*)d_in[7],  (const float*)d_in[11],
                          (const float*)d_in[15], (const float*)d_in[19] };
  const float* wtag   = (const float*)d_in[20];
  const float* btag   = (const float*)d_in[21];
  const float* trans  = (const float*)d_in[22];
  const float* startt = (const float*)d_in[23];
  const float* endt   = (const float*)d_in[24];
  float* out = (float*)d_out;

  // Workspace carve-up
  char* w = (char*)d_ws;
  size_t o = 0;
  auto alloc = [&](size_t bytes) { void* p = w + o; o = (o + bytes + 255) & ~(size_t)255; return p; };
  __bf16* x0    = (__bf16*)alloc((size_t)M_ * EMB_ * 2);
  __bf16* wihB[4], *whhB[4];
  const int inK[4] = { EMB_, EMB_, HID_, HID_ };
  for (int i = 0; i < 4; ++i) wihB[i] = (__bf16*)alloc((size_t)G4_ * inK[i] * 2);
  for (int i = 0; i < 4; ++i) whhB[i] = (__bf16*)alloc((size_t)G4_ * H_ * 2);
  __bf16* wtagB = (__bf16*)alloc((size_t)L_ * HID_ * 2);
  float* biasS[4];
  for (int i = 0; i < 4; ++i) biasS[i] = (float*)alloc((size_t)G4_ * 4);
  float*  xgF   = (float*) alloc((size_t)M_ * G4_ * 4);
  float*  xgB   = (float*) alloc((size_t)M_ * G4_ * 4);
  __bf16* act0  = (__bf16*)alloc((size_t)M_ * HID_ * 2);
  __bf16* act1  = (__bf16*)alloc((size_t)M_ * HID_ * 2);
  float*  emis  = (float*) alloc((size_t)M_ * L_ * 4);
  float*  logZ  = (float*) alloc(32 * 4);
  float*  gold  = (float*) alloc(32 * 4);
  int*    lastT = (int*)   alloc(32 * 4);
  unsigned char* bp = (unsigned char*)alloc((size_t)M_ * L_);

  // 1) Weight conversion to bf16 (+ bias sums)
  for (int i = 0; i < 4; ++i) {
    hipLaunchKernelGGL(k_f32_to_bf16, dim3(256), dim3(256), 0, stream,
                       wih[i], wihB[i], (long)G4_ * inK[i]);
    hipLaunchKernelGGL(k_f32_to_bf16, dim3(256), dim3(256), 0, stream,
                       whh[i], whhB[i], (long)G4_ * H_);
    hipLaunchKernelGGL(k_bias_sum, dim3(4), dim3(256), 0, stream,
                       bih[i], bhh[i], biasS[i], G4_);
  }
  hipLaunchKernelGGL(k_f32_to_bf16, dim3(64), dim3(256), 0, stream,
                     wtag, wtagB, (long)L_ * HID_);

  // 2) Embedding gather -> bf16 activations
  hipLaunchKernelGGL(k_embed, dim3(M_), dim3(EMB_), 0, stream, ids, embed, x0);

  const size_t REC_LDS = 32 * 1024 * 4 + 32 * 256 * 4 + 32 * 256 * 2; // 176KB

  // 3) Layer 0: input projections (WMMA GEMM, 32x64 macro-tiles), recurrence
  {
    const int blocks = (M_ / 32) * (G4_ / 64) / 8;   // 1024
    hipLaunchKernelGGL(HIP_KERNEL_NAME(k_wmma_gemm_bias<2, 4>),
                       dim3(blocks), dim3(256), 0, stream,
                       x0, wihB[0], biasS[0], xgF, M_, G4_, EMB_);
    hipLaunchKernelGGL(HIP_KERNEL_NAME(k_wmma_gemm_bias<2, 4>),
                       dim3(blocks), dim3(256), 0, stream,
                       x0, wihB[1], biasS[1], xgB, M_, G4_, EMB_);
    hipLaunchKernelGGL(k_lstm_rec, dim3(2), dim3(256), REC_LDS, stream,
                       xgF, xgB, whhB[0], whhB[1], act0, T_);
  }

  // 4) Layer 1
  {
    const int blocks = (M_ / 32) * (G4_ / 64) / 8;
    hipLaunchKernelGGL(HIP_KERNEL_NAME(k_wmma_gemm_bias<2, 4>),
                       dim3(blocks), dim3(256), 0, stream,
                       act0, wihB[2], biasS[2], xgF, M_, G4_, HID_);
    hipLaunchKernelGGL(HIP_KERNEL_NAME(k_wmma_gemm_bias<2, 4>),
                       dim3(blocks), dim3(256), 0, stream,
                       act0, wihB[3], biasS[3], xgB, M_, G4_, HID_);
    hipLaunchKernelGGL(k_lstm_rec, dim3(2), dim3(256), REC_LDS, stream,
                       xgF, xgB, whhB[2], whhB[3], act1, T_);
  }

  // 5) Emissions: act1 @ Wtag^T + b_tag (32x32 macro-tiles, N=32)
  {
    const int blocks = (M_ / 32) * (L_ / 32) / 8;    // 64
    hipLaunchKernelGGL(HIP_KERNEL_NAME(k_wmma_gemm_bias<2, 2>),
                       dim3(blocks), dim3(256), 0, stream,
                       act1, wtagB, btag, emis, M_, L_, HID_);
  }

  // 6) CRF: forward logZ, gold score, Viterbi, backtrack + loss
  hipLaunchKernelGGL(k_crf_forward, dim3(1), dim3(1024), 0, stream,
                     emis, mask, trans, startt, endt, logZ, T_);
  hipLaunchKernelGGL(k_gold, dim3(1), dim3(32), 0, stream,
                     emis, labels, mask, trans, startt, endt, gold, T_);
  hipLaunchKernelGGL(k_viterbi, dim3(1), dim3(1024), 0, stream,
                     emis, mask, trans, startt, endt, bp, lastT, T_);
  hipLaunchKernelGGL(k_backtrack_loss, dim3(1), dim3(32), 0, stream,
                     bp, lastT, logZ, gold, out, T_);
}
